// SAVSS2D_27152783246099
// MI455X (gfx1250) — matrix operand order repaired and checked by reading, not compiled, not run on hardware
//
#include <hip/hip_runtime.h>
#include <math.h>

// ---------------------------------------------------------------------------
// Types for CDNA5 WMMA (gfx1250, wave32)
// ---------------------------------------------------------------------------
typedef __attribute__((ext_vector_type(16))) __bf16 v16bf;
typedef __attribute__((ext_vector_type(8)))  float  v8f;

union FragBF {
  v16bf v;
  uint4 q[2];
};

__device__ __forceinline__ unsigned short f32_to_bf16(float f) {
  unsigned u = __float_as_uint(f);
  unsigned r = u + 0x7FFFu + ((u >> 16) & 1u);   // round-to-nearest-even
  return (unsigned short)(r >> 16);
}

// Fast sigmoid: v_exp_f32 + v_rcp_f32 instead of libm exp + precise divide.
__device__ __forceinline__ float sigmoid_fast(float v) {
  return __builtin_amdgcn_rcpf(1.0f + __expf(-v));
}

// ---------------------------------------------------------------------------
// Generic WMMA GEMM:  Y[b][o][p] = act( sum_c W[o][c] * Xbf[b][c][p] + bias[o] )
//   W  : f32 [M][128]   (row-major, K = C = 128 always)
//   Xbf: bf16 [B][128][HW]
//   Y  : f32 [B][M][HW]
// Block tile: 128 (M) x 64 (N), K=128. 256 threads = 8 wave32; each wave owns a
// 16-row M stripe and 4 accumulators covering the 64 N columns, reusing its A
// fragment across the 4 WMMAs per k-step (16 WMMA per wave total).
// grid: (HW/64, M/128, B)
// ---------------------------------------------------------------------------
#define LDSW 136   // padded row stride (bf16 units): 272B rows -> conflict-free
#define BN   64

__global__ __launch_bounds__(256) void gemm_bf16_wmma(
    const float* __restrict__ Wf,
    const unsigned short* __restrict__ Xbf,
    float* __restrict__ Y,
    const float* __restrict__ bias,
    int M, int HW, int sig_rows)
{
  __shared__ __align__(16) unsigned short lds_w[128 * LDSW];  // ~34.8 KB
  __shared__ __align__(16) unsigned short lds_x[BN  * LDSW];  // ~17.4 KB

  const int tid   = threadIdx.x;
  const int b     = blockIdx.z;
  const int obase = blockIdx.y * 128;
  const int p0    = blockIdx.x * BN;

  // Load + convert weight tile (128 out-rows x 128 K) to bf16 in LDS.
  for (int idx = tid; idx < 128 * 128; idx += 256) {
    int o = idx >> 7, c = idx & 127;
    lds_w[o * LDSW + c] = f32_to_bf16(Wf[(size_t)(obase + o) * 128 + c]);
  }
  // Load X tile transposed: lds_x[p][c]  (B-matrix wants rows of K per N).
  const unsigned short* xb = Xbf + (size_t)b * 128 * HW;
  for (int idx = tid; idx < 128 * BN; idx += 256) {
    int p = idx & (BN - 1), c = idx / BN;
    lds_x[p * LDSW + c] = xb[(size_t)c * HW + p0 + p];
  }
  __syncthreads();

  const int wave = tid >> 5;
  const int lane = tid & 31;
  const int m0   = wave * 16;       // 8 waves cover 128 M rows
  const int lrow = lane & 15;
  const int lhi  = lane >> 4;

  v8f acc[4] = {{}, {}, {}, {}};
  #pragma unroll
  for (int kk = 0; kk < 128; kk += 32) {
    // A (16x32, MxK): lane row m0+lrow; two contiguous 8-elt runs:
    //   i<8 : k = kk + i + 8*lhi ; i>=8 : k = kk + 16 + (i-8) + 8*lhi
    FragBF A;
    const unsigned short* ar = &lds_w[(m0 + lrow) * LDSW + kk + lhi * 8];
    A.q[0] = *(const uint4*)(ar);
    A.q[1] = *(const uint4*)(ar + 16);
    #pragma unroll
    for (int nt = 0; nt < 4; ++nt) {
      // B (32x16, KxN): lane col n=lrow; same k runs as A:
      //   i<8 : k = kk + i + 8*lhi ; i>=8 : k = kk + 16 + (i-8) + 8*lhi
      FragBF Bm;
      const unsigned short* br = &lds_x[(nt * 16 + lrow) * LDSW + kk + lhi * 8];
      Bm.q[0] = *(const uint4*)(br);
      Bm.q[1] = *(const uint4*)(br + 16);
      acc[nt] = __builtin_amdgcn_wmma_f32_16x16x32_bf16(
          /*neg_a=*/false, A.v, /*neg_b=*/false, Bm.v,
          /*c_mod=*/(short)0, acc[nt], /*reuse_a=*/false, /*reuse_b=*/false);
    }
  }

  // Epilogue: D layout — VGPR r: M = r + 8*lhi, N = lrow (+16*lhi offset none).
  float* yb = Y + ((size_t)b * M + obase) * HW;
  #pragma unroll
  for (int nt = 0; nt < 4; ++nt) {
    #pragma unroll
    for (int r = 0; r < 8; ++r) {
      int o = m0 + r + lhi * 8;
      float v = acc[nt][r];
      if (bias) v += bias[obase + o];
      if (obase + o < sig_rows) v = sigmoid_fast(v);
      yb[(size_t)o * HW + p0 + nt * 16 + lrow] = v;
    }
  }
}

// ---------------------------------------------------------------------------
// Elementwise / reduction helpers (fixed shape B=4, C=128, H=W=128)
// ---------------------------------------------------------------------------
__global__ void f32_to_bf16_kernel(const float* __restrict__ in,
                                   unsigned short* __restrict__ out, int n) {
  int i = blockIdx.x * blockDim.x + threadIdx.x;
  if (i < n) out[i] = f32_to_bf16(in[i]);
}

// One block per (b, group); 16 groups of 8 channels; reduce 8*16384 elems.
__global__ __launch_bounds__(256) void gn_stats_kernel(
    const float* __restrict__ x, float* __restrict__ stats) {
  const int bg = blockIdx.x;          // b*16 + g
  const int b = bg >> 4, g = bg & 15;
  const int HW = 16384;
  const float* base = x + ((size_t)b * 128 + g * 8) * HW;
  const int n = 8 * HW;
  float s = 0.f, ss = 0.f;
  for (int i = threadIdx.x; i < n; i += 256) {
    float v = base[i];
    s += v; ss += v * v;
  }
  __shared__ float sh_s[256], sh_q[256];
  sh_s[threadIdx.x] = s; sh_q[threadIdx.x] = ss;
  __syncthreads();
  for (int off = 128; off > 0; off >>= 1) {
    if (threadIdx.x < off) {
      sh_s[threadIdx.x] += sh_s[threadIdx.x + off];
      sh_q[threadIdx.x] += sh_q[threadIdx.x + off];
    }
    __syncthreads();
  }
  if (threadIdx.x == 0) {
    float mean = sh_s[0] / (float)n;
    float var  = sh_q[0] / (float)n - mean * mean;
    stats[bg * 2 + 0] = mean;
    stats[bg * 2 + 1] = rsqrtf(var + 1e-5f);
  }
}

// GroupNorm-apply + SiLU; optionally also emit a bf16 copy for the next GEMM.
__global__ void gn_apply_silu_kernel(
    const float* __restrict__ x, const float* __restrict__ stats,
    const float* __restrict__ gamma, const float* __restrict__ beta,
    float* __restrict__ out_f32, unsigned short* __restrict__ out_bf16, int n) {
  int i = blockIdx.x * blockDim.x + threadIdx.x;
  if (i >= n) return;
  int c = (i >> 14) & 127;      // HW = 2^14
  int b = i >> 21;              // C*HW = 2^21
  int g = c >> 3;
  float mean = stats[(b * 16 + g) * 2 + 0];
  float istd = stats[(b * 16 + g) * 2 + 1];
  float v = (x[i] - mean) * istd * gamma[c] + beta[c];
  float y = v * sigmoid_fast(v);
  out_f32[i] = y;
  if (out_bf16) out_bf16[i] = f32_to_bf16(y);
}

// Directional recurrence. One thread per (b, c, other) line, 128 steps.
//   h_t = a_t*h_{t-1} + b_t*x_t ; y_t = c_t*h_t + d_t*x_t
// coeff rows: [0,128)=a(sigmoid applied), [128,256)=b, [256,384)=c, [384,512)=d
// flip handled by scanning backwards (coeffs are pointwise).
__global__ __launch_bounds__(256) void scan_dir_kernel(
    const float* __restrict__ coeff,   // [B][512][HW]
    const float* __restrict__ xp,      // [B][128][HW]
    float* __restrict__ fused,         // [B][128][HW]
    int o_stride, int s_stride, int flip, int accumulate)
{
  const int id = blockIdx.x * 256 + threadIdx.x;   // 65536 threads
  const int other = id & 127;
  const int c     = (id >> 7) & 127;
  const int b     = id >> 14;
  const size_t HW = 16384;
  const size_t base = (size_t)other * o_stride;
  const float* pA = coeff + ((size_t)b * 512 + c) * HW + base;
  const float* pB = pA + (size_t)128 * HW;
  const float* pC = pA + (size_t)256 * HW;
  const float* pD = pA + (size_t)384 * HW;
  const float* pX = xp    + ((size_t)b * 128 + c) * HW + base;
  float*       pY = fused + ((size_t)b * 128 + c) * HW + base;

  float h = 0.f;
  const int t0 = flip ? 127 : 0;
  const int dt = flip ? -1 : 1;
  for (int s = 0; s < 128; ++s) {
    int t = t0 + s * dt;
    size_t off = (size_t)t * s_stride;
    if (s + 8 < 128) {
      // global_prefetch_b8: pull next cachelines (matters for the stride-512B
      // column scans; L2 is 192MB so this is an L2->WGP$ pull).
      size_t offp = (size_t)(t + 8 * dt) * s_stride;
      __builtin_prefetch(pA + offp, 0, 1);
      __builtin_prefetch(pB + offp, 0, 1);
      __builtin_prefetch(pC + offp, 0, 1);
      __builtin_prefetch(pD + offp, 0, 1);
      __builtin_prefetch(pX + offp, 0, 1);
    }
    float a  = pA[off], bb = pB[off], cc = pC[off], dd = pD[off];
    float xv = pX[off];
    h = a * h + bb * xv;
    float y = 0.25f * (cc * h + dd * xv);
    if (accumulate) pY[off] += y; else pY[off] = y;
  }
}

__global__ void mul_kernel(const float* __restrict__ a,
                           const float* __restrict__ b,
                           float* __restrict__ out, int n) {
  int i = blockIdx.x * blockDim.x + threadIdx.x;
  if (i < n) out[i] = a[i] * b[i];
}

// Depthwise 3x3, pad 1.
__global__ void dwconv3x3_kernel(const float* __restrict__ z,
                                 const float* __restrict__ k,
                                 float* __restrict__ out, int n) {
  int i = blockIdx.x * blockDim.x + threadIdx.x;
  if (i >= n) return;
  int w = i & 127, h = (i >> 7) & 127, c = (i >> 14) & 127, b = i >> 21;
  const float* zp = z + (((size_t)b * 128 + c) << 14);
  const float* kp = k + c * 9;
  float s = 0.f;
  #pragma unroll
  for (int dh = -1; dh <= 1; ++dh) {
    int hh = h + dh;
    if (hh < 0 || hh > 127) continue;
    #pragma unroll
    for (int dw = -1; dw <= 1; ++dw) {
      int ww = w + dw;
      if (ww < 0 || ww > 127) continue;
      s += zp[hh * 128 + ww] * kp[(dh + 1) * 3 + (dw + 1)];
    }
  }
  out[i] = s;
}

// ---------------------------------------------------------------------------
// Launch: full pipeline on `stream`.
// ---------------------------------------------------------------------------
extern "C" void kernel_launch(void* const* d_in, const int* in_sizes, int n_in,
                              void* d_out, int out_size, void* d_ws, size_t ws_size,
                              hipStream_t stream) {
  (void)in_sizes; (void)n_in; (void)out_size; (void)ws_size;

  const float* x      = (const float*)d_in[0];
  const float* w_in   = (const float*)d_in[1];
  const float* gn1_g  = (const float*)d_in[2];
  const float* gn1_b  = (const float*)d_in[3];
  const float* w_gate = (const float*)d_in[4];
  const float* b_gate = (const float*)d_in[5];
  const float* dyn_w  = (const float*)d_in[6];   // (4, 512, 128)
  const float* dyn_b  = (const float*)d_in[7];   // (4, 512)
  const float* dw_k   = (const float*)d_in[8];   // (128, 1, 3, 3)
  const float* w_out  = (const float*)d_in[9];
  const float* gn2_g  = (const float*)d_in[10];
  const float* gn2_b  = (const float*)d_in[11];
  float* out = (float*)d_out;

  const int N  = 4 * 128 * 128 * 128;   // 8,388,608
  const int HW = 128 * 128;

  char* ws = (char*)d_ws;
  const size_t MB = 1024 * 1024;
  float*          xp_raw = (float*)(ws);                    // 32MB (reused for z3)
  float*          xp     = (float*)(ws + 32  * MB);         // 32MB
  unsigned short* bfbuf  = (unsigned short*)(ws + 64 * MB); // 16MB (shared bf16 staging)
  float*          coeff  = (float*)(ws + 80  * MB);         // 128MB (per-direction)
  float*          fused  = (float*)(ws + 208 * MB);         // 32MB
  float*          gate   = (float*)(ws + 240 * MB);         // 32MB
  float*          zbuf   = (float*)(ws + 272 * MB);         // 32MB
  float*          z2buf  = (float*)(ws + 304 * MB);         // 32MB
  float*          stats  = (float*)(ws + 336 * MB);         // 128 floats

  dim3 blk(256);
  const int nbe = (N + 255) / 256;

  // in_proj: x -> bf16 -> WMMA GEMM -> GN1 -> SiLU (f32 + bf16 copies)
  f32_to_bf16_kernel<<<nbe, blk, 0, stream>>>(x, bfbuf, N);
  gemm_bf16_wmma<<<dim3(HW / BN, 1, 4), blk, 0, stream>>>(
      w_in, bfbuf, xp_raw, nullptr, 128, HW, 0);
  gn_stats_kernel<<<64, blk, 0, stream>>>(xp_raw, stats);
  gn_apply_silu_kernel<<<nbe, blk, 0, stream>>>(
      xp_raw, stats, gn1_g, gn1_b, xp, bfbuf, N);

  // 4-directional dynamic scans (coeff GEMM + recurrence, fused-sum in place)
  for (int dir = 0; dir < 4; ++dir) {
    gemm_bf16_wmma<<<dim3(HW / BN, 4, 4), blk, 0, stream>>>(
        dyn_w + (size_t)dir * 512 * 128, bfbuf, coeff,
        dyn_b + (size_t)dir * 512, 512, HW, /*sig_rows=*/128);
    const int axis3 = (dir < 2);                 // dirs 0,1: scan along W
    const int flip  = (dir == 1 || dir == 3);
    scan_dir_kernel<<<N / 128 / 256, blk, 0, stream>>>(
        coeff, xp, fused,
        /*o_stride=*/axis3 ? 128 : 1,
        /*s_stride=*/axis3 ? 1 : 128,
        flip, /*accumulate=*/dir != 0);
  }

  // gate = sigmoid(w_gate @ xp + b_gate); z = fused * gate
  gemm_bf16_wmma<<<dim3(HW / BN, 1, 4), blk, 0, stream>>>(
      w_gate, bfbuf, gate, b_gate, 128, HW, /*sig_rows=*/128);
  mul_kernel<<<nbe, blk, 0, stream>>>(fused, gate, zbuf, N);

  // depthwise 3x3 -> out_proj (WMMA) -> GN2 -> SiLU -> d_out
  dwconv3x3_kernel<<<nbe, blk, 0, stream>>>(zbuf, dw_k, z2buf, N);
  f32_to_bf16_kernel<<<nbe, blk, 0, stream>>>(z2buf, bfbuf, N);
  gemm_bf16_wmma<<<dim3(HW / BN, 1, 4), blk, 0, stream>>>(
      w_out, bfbuf, xp_raw, nullptr, 128, HW, 0);
  gn_stats_kernel<<<64, blk, 0, stream>>>(xp_raw, stats);
  gn_apply_silu_kernel<<<nbe, blk, 0, stream>>>(
      xp_raw, stats, gn2_g, gn2_b, out, nullptr, N);
}
